// MambaBlock_24893630447893
// MI455X (gfx1250) — matrix operand order repaired and checked by reading, not compiled, not run on hardware
//
#include <hip/hip_runtime.h>

#define D_MODEL 1024
#define D_INNER 2048
#define D_STATE 16
#define D_CONV  4
#define DT_RANK 64
#define BATCH   2
#define SEQ     2048
#define ROWS    (BATCH * SEQ)            // 4096
#define NXZ     (2 * D_INNER)            // 4096
#define NXP     (DT_RANK + 2 * D_STATE)  // 96 (actual)
#define NXP_PAD 128                      // padded to a multiple of 64

typedef __attribute__((ext_vector_type(16))) __bf16 v16bf;
typedef __attribute__((ext_vector_type(8)))  float  v8f;
typedef __attribute__((ext_vector_type(4)))  unsigned int u32x4;

__device__ __forceinline__ unsigned short f2bf(float f) {
  union { float f; unsigned u; } c; c.f = f;
  unsigned u = c.u;
  u += 0x7FFFu + ((u >> 16) & 1u);   // round-to-nearest-even
  return (unsigned short)(u >> 16);
}

// ---- cast f32 weight (K x N row-major) into bf16 WMMA-B fragment layout ----
// Output has Npad/16 tiles per 32-k chunk; columns in [N, Npad) are zero.
// tile t = (k/32)*(Npad/16) + n/16, 512 u16/tile,
//   lane = (n%16) + 16*((k%32)/16), elem = k%16 -> out[t*512 + lane*16 + elem]
__global__ void pack_b_bf16(const float* __restrict__ w, unsigned short* __restrict__ out,
                            int K, int N, int Npad) {
  int i = blockIdx.x * blockDim.x + threadIdx.x;
  int stride = gridDim.x * blockDim.x;
  int total = K * Npad;
  int ntiles = Npad >> 4;
  for (; i < total; i += stride) {
    int k = i / Npad, n = i % Npad;
    float val = (n < N) ? w[(size_t)k * N + n] : 0.f;
    int lane = (n & 15) + (((k & 31) >> 4) << 4);
    int e = k & 15;
    size_t o = ((size_t)(k >> 5) * ntiles + (n >> 4)) * 512 + lane * 16 + e;
    out[o] = f2bf(val);
  }
}

// ---------------- LayerNorm (f32 stats) -> bf16 ----------------
__global__ __launch_bounds__(256)
void layernorm_bf16(const float* __restrict__ x, const float* __restrict__ gamma,
                    const float* __restrict__ beta, unsigned short* __restrict__ xn) {
  __shared__ float red[256];
  const int row = blockIdx.x;
  const float* xr = x + (size_t)row * D_MODEL;
  float s = 0.f;
  for (int i = threadIdx.x; i < D_MODEL; i += 256) s += xr[i];
  red[threadIdx.x] = s; __syncthreads();
  for (int o = 128; o > 0; o >>= 1) {
    if (threadIdx.x < o) red[threadIdx.x] += red[threadIdx.x + o];
    __syncthreads();
  }
  const float mu = red[0] * (1.0f / D_MODEL);
  __syncthreads();
  float v = 0.f;
  for (int i = threadIdx.x; i < D_MODEL; i += 256) { float d = xr[i] - mu; v += d * d; }
  red[threadIdx.x] = v; __syncthreads();
  for (int o = 128; o > 0; o >>= 1) {
    if (threadIdx.x < o) red[threadIdx.x] += red[threadIdx.x + o];
    __syncthreads();
  }
  const float rs = rsqrtf(red[0] * (1.0f / D_MODEL) + 1e-5f);
  unsigned short* outr = xn + (size_t)row * D_MODEL;
  for (int i = threadIdx.x; i < D_MODEL; i += 256)
    outr[i] = f2bf((xr[i] - mu) * rs * gamma[i] + beta[i]);
}

// -------- bf16 WMMA GEMM: C(4096 x N) = A(row-major, lda) * Bpacked (+epilogue) ----
// Requires N % 64 == 0, K % 32 == 0, K >= 32, M % 128 == 0.
// EPI: 0 = none, 1 = softplus(acc + bias[col]), 2 = acc + resid[row*ldr+col]
// 128 threads (4 waves), block tile 128(M) x 64(N), BK=32; each wave owns a
// 32x64 output (2 M-subtiles x 4 N-subtiles = 8 WMMA/k-step, 4 shared B frags).
// A staged in LDS in fragment layout (32B contiguous per lane); B fragments
// read directly from packed global (L2-resident). Next A tile prefetched into
// registers while the current tile computes.
template <int EPI, bool STORE_BF16>
__global__ __launch_bounds__(128)
void gemm_wmma_bf16(const unsigned short* __restrict__ A, int lda,
                    const unsigned short* __restrict__ Bp,
                    float* __restrict__ C, unsigned short* __restrict__ Cbf, int ldc,
                    const float* __restrict__ bias,
                    const float* __restrict__ resid, int ldr,
                    int N, int K) {
  __shared__ __align__(32) unsigned short lA[8 * 512];  // 8 subtiles of 16x32
  const int tid  = threadIdx.x;
  const int wave = tid >> 5;
  const int lane = tid & 31;
  const int ln   = lane & 15;
  const int lg   = lane >> 4;
  const int row0 = blockIdx.y * 128;
  const int col0 = blockIdx.x * 64;
  const int ntiles = N >> 4;
  const int nt0 = col0 >> 4;

  // A-staging addresses: 512 16-byte chunks (128 rows x 4 k-chunks), 4/thread.
  // chunk ci -> row r = ci/4, kchunk c = ci%4 ; g = c>>1, e0 = (c&1)*8
  int rr[4], cc[4], ldst[4];
#pragma unroll
  for (int it = 0; it < 4; ++it) {
    int ci = tid + it * 128;
    int r = ci >> 2, c = ci & 3;
    int g = c >> 1, e0 = (c & 1) << 3;
    rr[it] = r; cc[it] = c;
    ldst[it] = ((r >> 4) << 9) + (((r & 15) + (g << 4)) << 4) + e0;
  }

  v8f acc[2][4] = {};
  u32x4 ra[4];

  // preload k-chunk 0
#pragma unroll
  for (int it = 0; it < 4; ++it)
    ra[it] = *reinterpret_cast<const u32x4*>(&A[(size_t)(row0 + rr[it]) * lda + cc[it] * 8]);

  int k0 = 0;
  for (; k0 + 32 < K; k0 += 32) {
#pragma unroll
    for (int it = 0; it < 4; ++it)
      *reinterpret_cast<u32x4*>(&lA[ldst[it]]) = ra[it];
    __syncthreads();

    // prefetch next A tile while this one computes
#pragma unroll
    for (int it = 0; it < 4; ++it)
      ra[it] = *reinterpret_cast<const u32x4*>(
          &A[(size_t)(row0 + rr[it]) * lda + (k0 + 32) + cc[it] * 8]);

    v16bf av0 = *reinterpret_cast<const v16bf*>(&lA[(wave << 9) + (lane << 4)]);
    v16bf av1 = *reinterpret_cast<const v16bf*>(&lA[((wave + 4) << 9) + (lane << 4)]);
    const size_t kbase = (size_t)(k0 >> 5) * ntiles;
    v16bf bv[4];
#pragma unroll
    for (int ns = 0; ns < 4; ++ns)
      bv[ns] = *reinterpret_cast<const v16bf*>(&Bp[((kbase + nt0 + ns) << 9) + (lane << 4)]);
#pragma unroll
    for (int ns = 0; ns < 4; ++ns) {
      acc[0][ns] = __builtin_amdgcn_wmma_f32_16x16x32_bf16(
          false, av0, false, bv[ns], (short)0, acc[0][ns], false, false);
      acc[1][ns] = __builtin_amdgcn_wmma_f32_16x16x32_bf16(
          false, av1, false, bv[ns], (short)0, acc[1][ns], false, false);
    }
    __syncthreads();
  }

  // last k-chunk (peeled: no prefetch, no trailing barrier)
  {
#pragma unroll
    for (int it = 0; it < 4; ++it)
      *reinterpret_cast<u32x4*>(&lA[ldst[it]]) = ra[it];
    __syncthreads();
    v16bf av0 = *reinterpret_cast<const v16bf*>(&lA[(wave << 9) + (lane << 4)]);
    v16bf av1 = *reinterpret_cast<const v16bf*>(&lA[((wave + 4) << 9) + (lane << 4)]);
    const size_t kbase = (size_t)(k0 >> 5) * ntiles;
    v16bf bv[4];
#pragma unroll
    for (int ns = 0; ns < 4; ++ns)
      bv[ns] = *reinterpret_cast<const v16bf*>(&Bp[((kbase + nt0 + ns) << 9) + (lane << 4)]);
#pragma unroll
    for (int ns = 0; ns < 4; ++ns) {
      acc[0][ns] = __builtin_amdgcn_wmma_f32_16x16x32_bf16(
          false, av0, false, bv[ns], (short)0, acc[0][ns], false, false);
      acc[1][ns] = __builtin_amdgcn_wmma_f32_16x16x32_bf16(
          false, av1, false, bv[ns], (short)0, acc[1][ns], false, false);
    }
  }

  // store: C/D layout — VGPR r: M = r + 8*lanehalf, N = lane%16
#pragma unroll
  for (int ms = 0; ms < 2; ++ms) {
#pragma unroll
    for (int ns = 0; ns < 4; ++ns) {
#pragma unroll
      for (int r = 0; r < 8; ++r) {
        int gm = row0 + ms * 64 + wave * 16 + r + lg * 8;
        int gn = (nt0 + ns) * 16 + ln;
        float v = acc[ms][ns][r];
        if constexpr (EPI == 1) {
          v += bias[gn];
          v = (v > 20.f) ? v : __logf(1.f + __expf(v));  // softplus
        }
        if constexpr (EPI == 2) {
          v += resid[(size_t)gm * ldr + gn];
        }
        C[(size_t)gm * ldc + gn] = v;
        if constexpr (STORE_BF16) Cbf[(size_t)gm * ldc + gn] = f2bf(v);
      }
    }
  }
}

// ---------------- depthwise causal conv(4) + SiLU ----------------
__global__ void conv_silu(const float* __restrict__ xz, const float* __restrict__ conv_w,
                          const float* __restrict__ conv_b,
                          float* __restrict__ xb, unsigned short* __restrict__ xb_bf) {
  size_t idx = (size_t)blockIdx.x * blockDim.x + threadIdx.x;
  const size_t total = (size_t)ROWS * D_INNER;
  if (idx >= total) return;
  const int d = (int)(idx % D_INNER);
  const size_t row = idx / D_INNER;
  const int t = (int)(row % SEQ);
  const size_t rb = row - (size_t)t;  // first row of this batch
  float acc = conv_b[d];
#pragma unroll
  for (int k = 0; k < D_CONV; ++k) {
    int tt = t + k - (D_CONV - 1);
    if (tt >= 0) acc += conv_w[d * D_CONV + k] * xz[(rb + (size_t)tt) * NXZ + d];
  }
  float sv = acc / (1.f + __expf(-acc));  // SiLU
  xb[idx]    = sv;
  xb_bf[idx] = f2bf(sv);
}

// ---------------- selective scan: one thread per (batch, channel) ----------------
__global__ __launch_bounds__(256)
void selective_scan(const float* __restrict__ xz, const float* __restrict__ xb,
                    const float* __restrict__ dt, const float* __restrict__ xp,
                    const float* __restrict__ A_log, const float* __restrict__ D_param,
                    unsigned short* __restrict__ y_bf) {
  const int chunks = D_INNER / 256;
  const int b = blockIdx.x / chunks;
  const int d = (blockIdx.x % chunks) * 256 + threadIdx.x;
  float Ad[D_STATE], h[D_STATE];
#pragma unroll
  for (int s = 0; s < D_STATE; ++s) {
    Ad[s] = -__expf(A_log[d * D_STATE + s]);
    h[s]  = 0.f;
  }
  const float Dp = D_param[d];
  for (int t = 0; t < SEQ; ++t) {
    const size_t row = (size_t)b * SEQ + t;
    const float dtv = dt[row * D_INNER + d];
    const float xv  = xb[row * D_INNER + d];
    const float* prow = xp + row * NXP_PAD;
    const float dx = dtv * xv;
    float y = 0.f;
#pragma unroll
    for (int s = 0; s < D_STATE; ++s) {
      h[s] = __expf(dtv * Ad[s]) * h[s] + dx * prow[DT_RANK + s];
      y += h[s] * prow[DT_RANK + D_STATE + s];
    }
    y += xv * Dp;
    const float z = xz[row * NXZ + D_INNER + d];
    y *= z / (1.f + __expf(-z));  // * silu(z)
    y_bf[row * D_INNER + d] = f2bf(y);
  }
}

// ---------------- host launcher ----------------
extern "C" void kernel_launch(void* const* d_in, const int* in_sizes, int n_in,
                              void* d_out, int out_size, void* d_ws, size_t ws_size,
                              hipStream_t stream) {
  (void)in_sizes; (void)n_in; (void)out_size; (void)ws_size;
  const float* x       = (const float*)d_in[0];
  const float* gamma   = (const float*)d_in[1];
  const float* beta    = (const float*)d_in[2];
  const float* w_in    = (const float*)d_in[3];
  const float* conv_w  = (const float*)d_in[4];
  const float* conv_b  = (const float*)d_in[5];
  const float* w_x     = (const float*)d_in[6];
  const float* w_dt    = (const float*)d_in[7];
  const float* b_dt    = (const float*)d_in[8];
  const float* A_log   = (const float*)d_in[9];
  const float* D_param = (const float*)d_in[10];
  const float* w_out   = (const float*)d_in[11];
  float* out = (float*)d_out;

  char* ws = (char*)d_ws;
  size_t off = 0;
  auto alloc = [&](size_t bytes) -> void* {
    void* p = ws + off;
    off += (bytes + 255) & ~(size_t)255;
    return p;
  };

  unsigned short* pw_in  = (unsigned short*)alloc((size_t)D_MODEL * NXZ * 2);
  unsigned short* pw_out = (unsigned short*)alloc((size_t)D_INNER * D_MODEL * 2);
  unsigned short* pw_x   = (unsigned short*)alloc((size_t)D_INNER * NXP_PAD * 2);
  unsigned short* pw_dt  = (unsigned short*)alloc((size_t)DT_RANK * D_INNER * 2);
  unsigned short* xn_bf  = (unsigned short*)alloc((size_t)ROWS * D_MODEL * 2);
  float*          xz     = (float*)alloc((size_t)ROWS * NXZ * 4);
  float*          xb     = (float*)alloc((size_t)ROWS * D_INNER * 4);
  unsigned short* xb_bf  = (unsigned short*)alloc((size_t)ROWS * D_INNER * 2);
  float*          xp     = (float*)alloc((size_t)ROWS * NXP_PAD * 4);
  unsigned short* xp_bf  = (unsigned short*)alloc((size_t)ROWS * NXP_PAD * 2);
  float*          dtb    = (float*)alloc((size_t)ROWS * D_INNER * 4);
  unsigned short* y_bf   = (unsigned short*)alloc((size_t)ROWS * D_INNER * 2);

  auto pack = [&](const float* src, unsigned short* dst, int K, int N, int Npad) {
    int n = K * Npad;
    int blocks = (n + 255) / 256; if (blocks > 4096) blocks = 4096;
    pack_b_bf16<<<blocks, 256, 0, stream>>>(src, dst, K, N, Npad);
  };
  pack(w_in,  pw_in,  D_MODEL, NXZ,     NXZ);
  pack(w_out, pw_out, D_INNER, D_MODEL, D_MODEL);
  pack(w_x,   pw_x,   D_INNER, NXP,     NXP_PAD);
  pack(w_dt,  pw_dt,  DT_RANK, D_INNER, D_INNER);

  layernorm_bf16<<<ROWS, 256, 0, stream>>>(x, gamma, beta, xn_bf);

  // GEMM1: xz = xn @ w_in  (4096 x 4096, K=1024)
  gemm_wmma_bf16<0, false><<<dim3(NXZ / 64, ROWS / 128), 128, 0, stream>>>(
      xn_bf, D_MODEL, pw_in, xz, nullptr, NXZ,
      nullptr, nullptr, 0, NXZ, D_MODEL);

  // conv + SiLU on x_b half of xz
  {
    size_t total = (size_t)ROWS * D_INNER;
    conv_silu<<<(unsigned)((total + 255) / 256), 256, 0, stream>>>(
        xz, conv_w, conv_b, xb, xb_bf);
  }

  // GEMM2: xp = x_b @ w_x  (4096 x 128pad, K=2048), keep bf16 copy for GEMM3
  gemm_wmma_bf16<0, true><<<dim3(NXP_PAD / 64, ROWS / 128), 128, 0, stream>>>(
      xb_bf, D_INNER, pw_x, xp, xp_bf, NXP_PAD,
      nullptr, nullptr, 0, NXP_PAD, D_INNER);

  // GEMM3: dt = softplus(dt_raw @ w_dt + b_dt)  (4096 x 2048, K=64)
  gemm_wmma_bf16<1, false><<<dim3(D_INNER / 64, ROWS / 128), 128, 0, stream>>>(
      xp_bf, NXP_PAD, pw_dt, dtb, nullptr, D_INNER,
      b_dt, nullptr, 0, D_INNER, DT_RANK);

  // selective scan + D skip + gate
  selective_scan<<<BATCH * (D_INNER / 256), 256, 0, stream>>>(
      xz, xb, dtb, xp, A_log, D_param, y_bf);

  // GEMM4: out = y @ w_out + residual  (4096 x 1024, K=2048)
  gemm_wmma_bf16<2, false><<<dim3(D_MODEL / 64, ROWS / 128), 128, 0, stream>>>(
      y_bf, D_INNER, pw_out, out, nullptr, D_MODEL,
      nullptr, x, D_MODEL, D_MODEL, D_INNER);
}
